// SFGCN_15582141350528
// MI455X (gfx1250) — matrix operand order repaired and checked by reading, not compiled, not run on hardware
//
#include <hip/hip_runtime.h>

typedef __attribute__((ext_vector_type(16))) _Float16 v16h;
typedef __attribute__((ext_vector_type(8)))  _Float16 v8h;
typedef __attribute__((ext_vector_type(8)))  float    v8f;

#define NNODE 50000
#define NROW  250000   // NNODE*5
#define NEDGE 1600000

static __device__ __forceinline__ v8f wmma_f16(v16h a, v16h b, v8f c) {
    return __builtin_amdgcn_wmma_f32_16x16x32_f16(false, a, false, b, (short)0, c, false, false);
}

// Build A-matrix fragment (16x32 f16) for this lane from a row pointer at k0.
// Lane holds row m=lane&15; elements 0..15 = K(half*16 .. +16), contiguous (matches B loads)
static __device__ __forceinline__ v16h load_a(const _Float16* rowk0, int hh) {
    v8h lo = *(const v8h*)(rowk0 + hh * 16);
    v8h hi = *(const v8h*)(rowk0 + hh * 16 + 8);
    v16h a;
#pragma unroll
    for (int i = 0; i < 8; ++i) { a[i] = lo[i]; a[8 + i] = hi[i]; }
    return a;
}

// ---------------- weight prep: f32 -> padded f16 B-layouts ----------------
__global__ void prep_kernel(const float* __restrict__ fcnn_w, const float* __restrict__ lin1_w,
                            const float* __restrict__ wih,    const float* __restrict__ whh,
                            const float* __restrict__ w1,     const float* __restrict__ wc,
                            const float* __restrict__ w2,
                            _Float16* fcnnW16, _Float16* lin1W16, _Float16* wih16,
                            _Float16* whh16, _Float16* gcnW16) {
    for (int idx = blockIdx.x * blockDim.x + threadIdx.x; idx < 80896;
         idx += gridDim.x * blockDim.x) {
        int i = idx;
        if (i < 33280) {                       // fcnn_w (80,392) -> (80,416)
            int n = i / 416, k = i % 416;
            fcnnW16[i] = (_Float16)(k < 392 ? fcnn_w[n * 392 + k] : 0.0f);
            continue;
        }
        i -= 33280;
        if (i < 4608) {                        // lin1_w (40,80) -> (48,96)
            int n = i / 96, k = i % 96;
            lin1W16[i] = (_Float16)((n < 40 && k < 80) ? lin1_w[n * 80 + k] : 0.0f);
            continue;
        }
        i -= 4608;
        if (i < 12288) {                       // gru_wih (192,41) -> (192,64)
            int n = i / 64, k = i % 64;
            wih16[i] = (_Float16)(k < 41 ? wih[n * 41 + k] : 0.0f);
            continue;
        }
        i -= 12288;
        if (i < 12288) {                       // gru_whh (192,64)
            whh16[i] = (_Float16)whh[i];
            continue;
        }
        i -= 12288;
        {                                      // gcn W (104,40) -> Bt (48,128) x3
            int s = i / 6144, rem = i % 6144, n = rem / 128, k = rem % 128;
            const float* W = (s == 0) ? w1 : ((s == 1) ? wc : w2);
            gcnW16[i] = (_Float16)((n < 40 && k < 104) ? W[k * 40 + n] : 0.0f);
        }
    }
}

// ------------- fused conv1d + fcnn(392->80) + lin1(80->40), WMMA -------------
__global__ __launch_bounds__(32) void fc_kernel(const float* __restrict__ x,
                                                const float* __restrict__ conv_w,
                                                const float* __restrict__ conv_b,
                                                const _Float16* __restrict__ fcnnW16,
                                                const float* __restrict__ fcnn_b,
                                                const _Float16* __restrict__ lin1W16,
                                                const float* __restrict__ lin1_b,
                                                _Float16* __restrict__ xs16) {
    __shared__ __align__(32) _Float16 feat[16 * 416];
    __shared__ __align__(32) _Float16 mid[16 * 96];
    const int tile = blockIdx.x;       // 15625 tiles of 16 rows
    const int lane = threadIdx.x;
    const int r0 = tile * 16;
    // phase 1: conv(1->8,k8,s8)+relu into LDS, zero pad K to 416
    for (int idx = lane; idx < 16 * 416; idx += 32) {
        int row = idx / 416, c = idx % 416;
        float val = 0.0f;
        if (c < 392) {
            int ch = c / 49, p = c % 49;
            const float* xp = x + (size_t)(r0 + row) * 395 + 3 + p * 8;
            val = conv_b[ch];
#pragma unroll
            for (int j = 0; j < 8; ++j) val += xp[j] * conv_w[ch * 8 + j];
            val = val > 0.0f ? val : 0.0f;
        }
        feat[idx] = (_Float16)val;
    }
    const int m = lane & 15, hh = lane >> 4;
    // phase 2: fcnn GEMM 16x416 @ B(80x416)
    v8f acc[5];
#pragma unroll
    for (int t = 0; t < 5; ++t) acc[t] = (v8f)0.0f;
    for (int kc = 0; kc < 13; ++kc) {
        const int k0 = kc * 32;
        v16h a = load_a(&feat[m * 416 + k0], hh);
#pragma unroll
        for (int nt = 0; nt < 5; ++nt) {
            v16h b = *(const v16h*)&fcnnW16[(nt * 16 + m) * 416 + k0 + hh * 16];
            acc[nt] = wmma_f16(a, b, acc[nt]);
        }
    }
    // bias+relu -> mid LDS (16x96), pad cols 80..95
#pragma unroll
    for (int nt = 0; nt < 5; ++nt) {
        int n = nt * 16 + m;
        float bia = fcnn_b[n];
#pragma unroll
        for (int r = 0; r < 8; ++r) {
            float f = acc[nt][r] + bia;
            mid[(r + hh * 8) * 96 + n] = (_Float16)(f > 0.0f ? f : 0.0f);
        }
    }
#pragma unroll
    for (int r = 0; r < 8; ++r) mid[(r + hh * 8) * 96 + 80 + m] = (_Float16)0.0f;
    // phase 3: lin1 GEMM 16x96 @ B(48x96)
    v8f acc2[3];
#pragma unroll
    for (int t = 0; t < 3; ++t) acc2[t] = (v8f)0.0f;
    for (int kc = 0; kc < 3; ++kc) {
        const int k0 = kc * 32;
        v16h a = load_a(&mid[m * 96 + k0], hh);
#pragma unroll
        for (int nt = 0; nt < 3; ++nt) {
            v16h b = *(const v16h*)&lin1W16[(nt * 16 + m) * 96 + k0 + hh * 16];
            acc2[nt] = wmma_f16(a, b, acc2[nt]);
        }
    }
#pragma unroll
    for (int nt = 0; nt < 3; ++nt) {
        int n = nt * 16 + m;
        if (n < 40) {
            float bia = lin1_b[n];
#pragma unroll
            for (int r = 0; r < 8; ++r) {
                float f = acc2[nt][r] + bia;
                xs16[(size_t)(r0 + r + hh * 8) * 64 + n] = (_Float16)(f > 0.0f ? f : 0.0f);
            }
        }
    }
}

// xs col 40 = y = x[:,:,2]; cols 41..63 = 0
__global__ void y_kernel(const float* __restrict__ x, _Float16* __restrict__ xs16) {
    int r = blockIdx.x * blockDim.x + threadIdx.x;
    if (r >= NROW) return;
    _Float16* row = xs16 + (size_t)r * 64;
    row[40] = (_Float16)x[(size_t)r * 395 + 2];
#pragma unroll
    for (int c = 41; c < 64; ++c) row[c] = (_Float16)0.0f;
}

// ---------------- GRU: 4 steps, WMMA per 16-node tile ----------------
__global__ __launch_bounds__(32) void gru_kernel(const _Float16* __restrict__ xs16,
                                                 const _Float16* __restrict__ wih16,
                                                 const _Float16* __restrict__ whh16,
                                                 const float* __restrict__ bih,
                                                 const float* __restrict__ bhh,
                                                 _Float16* __restrict__ h16) {
    __shared__ __align__(32) _Float16 hbuf[16 * 64];
    const int tile = blockIdx.x;   // 3125 tiles
    const int lane = threadIdx.x;
    const int m = lane & 15, hh = lane >> 4;
    const int node0 = tile * 16;
    v8f hD[4];
#pragma unroll
    for (int j = 0; j < 4; ++j) hD[j] = (v8f)0.0f;
    for (int idx = lane; idx < 16 * 64; idx += 32) hbuf[idx] = (_Float16)0.0f;
    for (int t = 0; t < 4; ++t) {
        v16h ax[2], ah[2];
#pragma unroll
        for (int kc = 0; kc < 2; ++kc) {
            ax[kc] = load_a(xs16 + ((size_t)(node0 + m) * 5 + t) * 64 + kc * 32, hh);
            ah[kc] = load_a(&hbuf[m * 64 + kc * 32], hh);
        }
#pragma unroll
        for (int j = 0; j < 4; ++j) {
            v8f aR = (v8f)0.0f, aZ = (v8f)0.0f, aC = (v8f)0.0f, aHC = (v8f)0.0f;
            const int nR = j * 16 + m, nZ = 64 + j * 16 + m, nC = 128 + j * 16 + m;
#pragma unroll
            for (int kc = 0; kc < 2; ++kc) {
                const int ko = kc * 32 + hh * 16;
                aR = wmma_f16(ax[kc], *(const v16h*)&wih16[nR * 64 + ko], aR);
                aR = wmma_f16(ah[kc], *(const v16h*)&whh16[nR * 64 + ko], aR);
                aZ = wmma_f16(ax[kc], *(const v16h*)&wih16[nZ * 64 + ko], aZ);
                aZ = wmma_f16(ah[kc], *(const v16h*)&whh16[nZ * 64 + ko], aZ);
                aC = wmma_f16(ax[kc], *(const v16h*)&wih16[nC * 64 + ko], aC);
                aHC = wmma_f16(ah[kc], *(const v16h*)&whh16[nC * 64 + ko], aHC);
            }
            const float bRs = bih[nR] + bhh[nR];
            const float bZs = bih[nZ] + bhh[nZ];
            const float bCi = bih[nC], bCh = bhh[nC];
            v8f hn;
#pragma unroll
            for (int r = 0; r < 8; ++r) {
                float rg = 1.0f / (1.0f + __expf(-(aR[r] + bRs)));
                float zg = 1.0f / (1.0f + __expf(-(aZ[r] + bZs)));
                float cg = tanhf(aC[r] + bCi + rg * (aHC[r] + bCh));
                hn[r] = (1.0f - zg) * cg + zg * hD[j][r];
            }
            hD[j] = hn;
#pragma unroll
            for (int r = 0; r < 8; ++r)
                hbuf[(r + hh * 8) * 64 + j * 16 + m] = (_Float16)hn[r];
        }
    }
    // hT -> h16 cols 40..103
#pragma unroll
    for (int j = 0; j < 4; ++j)
#pragma unroll
        for (int r = 0; r < 8; ++r)
            h16[(size_t)(node0 + r + hh * 8) * 128 + 40 + j * 16 + m] = (_Float16)hD[j][r];
}

// h16 cols 0..39 = xs[:,4,:40]; cols 104..127 = 0
__global__ void h0_kernel(const _Float16* __restrict__ xs16, _Float16* __restrict__ h16) {
    int idx = blockIdx.x * blockDim.x + threadIdx.x;
    if (idx >= NNODE * 64) return;
    int node = idx / 64, c = idx % 64;
    if (c < 40)
        h16[(size_t)node * 128 + c] = xs16[((size_t)node * 5 + 4) * 64 + c];
    else
        h16[(size_t)node * 128 + c + 64] = (_Float16)0.0f;   // 104..127
}

// ---------------- GCN projection: xw = h16 @ W  (3 weight sets) ----------------
__global__ __launch_bounds__(32) void xw_kernel(const _Float16* __restrict__ h16,
                                                const _Float16* __restrict__ gcnW16,
                                                float* __restrict__ xw) {
    const int tile = blockIdx.x, set = blockIdx.y;
    const int lane = threadIdx.x, m = lane & 15, hh = lane >> 4;
    const _Float16* B = gcnW16 + (size_t)set * 48 * 128;
    float* out = xw + (size_t)set * NNODE * 40;
    v8f acc[3];
#pragma unroll
    for (int t = 0; t < 3; ++t) acc[t] = (v8f)0.0f;
    for (int kc = 0; kc < 4; ++kc) {
        const int k0 = kc * 32;
        v16h a = load_a(h16 + (size_t)(tile * 16 + m) * 128 + k0, hh);
#pragma unroll
        for (int nt = 0; nt < 3; ++nt) {
            v16h b = *(const v16h*)&B[(nt * 16 + m) * 128 + k0 + hh * 16];
            acc[nt] = wmma_f16(a, b, acc[nt]);
        }
    }
#pragma unroll
    for (int nt = 0; nt < 3; ++nt) {
        int n = nt * 16 + m;
        if (n < 40) {
#pragma unroll
            for (int r = 0; r < 8; ++r)
                out[(size_t)(tile * 16 + r + hh * 8) * 40 + n] = acc[nt][r];
        }
    }
}

// ---------------- degrees / norm ----------------
__global__ void fill_kernel(float* p, float v, int n) {
    int i = blockIdx.x * blockDim.x + threadIdx.x;
    if (i < n) p[i] = v;
}
__global__ void deg_edge_kernel(const int* __restrict__ dst, float* __restrict__ deg) {
    int e = blockIdx.x * blockDim.x + threadIdx.x;
    if (e < NEDGE) atomicAdd(&deg[dst[e]], 1.0f);
}
__global__ void dinv_kernel(const float* __restrict__ deg, float* __restrict__ dinv, int n) {
    int i = blockIdx.x * blockDim.x + threadIdx.x;
    if (i < n) { float d = deg[i]; dinv[i] = d > 0.0f ? rsqrtf(d) : 0.0f; }
}

// acc init with self-loop term: acc = xw * dinv^2
__global__ void acc_init_kernel(const float* __restrict__ xw, const float* __restrict__ dinv,
                                float* __restrict__ acc) {
    int idx = blockIdx.x * blockDim.x + threadIdx.x;
    if (idx >= 4 * NNODE * 40) return;
    int which = idx / (NNODE * 40);
    int rem = idx % (NNODE * 40);
    int node = rem / 40, c = rem % 40;
    const int xsel[4] = {0, 1, 1, 2};   // emb1<-sgcn1, com1<-cgcn, com2<-cgcn, emb2<-sgcn2
    const int dsel[4] = {0, 0, 1, 1};
    float d = dinv[dsel[which] * NNODE + node];
    acc[idx] = xw[(size_t)xsel[which] * NNODE * 40 + (size_t)node * 40 + c] * d * d;
}

// per-edge scatter into two accumulators (shares index/norm reads)
__global__ void edge_agg_kernel(const int* __restrict__ src, const int* __restrict__ dst,
                                const float* __restrict__ dinv,
                                const float* __restrict__ xwA, const float* __restrict__ xwB,
                                float* __restrict__ accA, float* __restrict__ accB) {
    int e = blockIdx.x * blockDim.x + threadIdx.x;
    if (e >= NEDGE) return;
    int s = src[e], d = dst[e];
    float nrm = dinv[s] * dinv[d];
    const float4* xa = (const float4*)(xwA + (size_t)s * 40);
    const float4* xb = (const float4*)(xwB + (size_t)s * 40);
    float* aa = accA + (size_t)d * 40;
    float* ab = accB + (size_t)d * 40;
#pragma unroll
    for (int q = 0; q < 10; ++q) {
        float4 va = xa[q], vb = xb[q];
        atomicAdd(&aa[q * 4 + 0], va.x * nrm); atomicAdd(&aa[q * 4 + 1], va.y * nrm);
        atomicAdd(&aa[q * 4 + 2], va.z * nrm); atomicAdd(&aa[q * 4 + 3], va.w * nrm);
        atomicAdd(&ab[q * 4 + 0], vb.x * nrm); atomicAdd(&ab[q * 4 + 1], vb.y * nrm);
        atomicAdd(&ab[q * 4 + 2], vb.z * nrm); atomicAdd(&ab[q * 4 + 3], vb.w * nrm);
    }
}

// bias + relu -> d_out regions (emb1, com1, com2, emb2)
__global__ void bias_relu_kernel(const float* __restrict__ acc,
                                 const float* __restrict__ b1, const float* __restrict__ bc,
                                 const float* __restrict__ b2, float* __restrict__ out_emb) {
    int idx = blockIdx.x * blockDim.x + threadIdx.x;
    if (idx >= 4 * NNODE * 40) return;
    int which = idx / (NNODE * 40);
    int c = idx % 40;
    const float* b = (which == 0) ? b1 : ((which == 3) ? b2 : bc);
    float v = acc[idx] + b[c];
    out_emb[idx] = v > 0.0f ? v : 0.0f;   // contiguous emb1|com1|com2|emb2
}

// ---------------- attention + MLP ----------------
__global__ void att_kernel(const float* __restrict__ emb1, const float* __restrict__ com1,
                           const float* __restrict__ com2, const float* __restrict__ emb2,
                           const float* __restrict__ att1_w, const float* __restrict__ att1_b,
                           const float* __restrict__ att2_w, const float* __restrict__ mlp_w,
                           const float* __restrict__ mlp_b,
                           float* __restrict__ output, float* __restrict__ beta,
                           float* __restrict__ emb) {
    int i = blockIdx.x * blockDim.x + threadIdx.x;
    if (i >= NNODE) return;
    float z[3][40];
#pragma unroll
    for (int c = 0; c < 40; ++c) {
        z[0][c] = emb1[(size_t)i * 40 + c];
        z[1][c] = emb2[(size_t)i * 40 + c];
        z[2][c] = 0.5f * (com1[(size_t)i * 40 + c] + com2[(size_t)i * 40 + c]);
    }
    float w[3];
#pragma unroll
    for (int s = 0; s < 3; ++s) {
        float a = 0.0f;
        for (int j = 0; j < 16; ++j) {
            float d = att1_b[j];
#pragma unroll
            for (int c = 0; c < 40; ++c) d += z[s][c] * att1_w[j * 40 + c];
            a += att2_w[j] * tanhf(d);
        }
        w[s] = a;
    }
    float mx = fmaxf(w[0], fmaxf(w[1], w[2]));
    float e0 = __expf(w[0] - mx), e1 = __expf(w[1] - mx), e2 = __expf(w[2] - mx);
    float inv = 1.0f / (e0 + e1 + e2);
    float b0 = e0 * inv, b1 = e1 * inv, b2 = e2 * inv;
    beta[(size_t)i * 3 + 0] = b0; beta[(size_t)i * 3 + 1] = b1; beta[(size_t)i * 3 + 2] = b2;
    float o = mlp_b[0];
#pragma unroll
    for (int c = 0; c < 40; ++c) {
        float ec = b0 * z[0][c] + b1 * z[1][c] + b2 * z[2][c];
        emb[(size_t)i * 40 + c] = ec;
        o += ec * mlp_w[c];
    }
    output[i] = o;
}

extern "C" void kernel_launch(void* const* d_in, const int* in_sizes, int n_in,
                              void* d_out, int out_size, void* d_ws, size_t ws_size,
                              hipStream_t stream) {
    const float* x        = (const float*)d_in[0];
    const int*   ei       = (const int*)d_in[1];
    const int*   fei      = (const int*)d_in[2];
    const float* conv_w   = (const float*)d_in[3];
    const float* conv_b   = (const float*)d_in[4];
    const float* fcnn_w   = (const float*)d_in[5];
    const float* fcnn_b   = (const float*)d_in[6];
    const float* lin1_w   = (const float*)d_in[7];
    const float* lin1_b   = (const float*)d_in[8];
    const float* gru_wih  = (const float*)d_in[9];
    const float* gru_whh  = (const float*)d_in[10];
    const float* gru_bih  = (const float*)d_in[11];
    const float* gru_bhh  = (const float*)d_in[12];
    const float* sgcn1_w  = (const float*)d_in[13];
    const float* sgcn1_b  = (const float*)d_in[14];
    const float* sgcn2_w  = (const float*)d_in[15];
    const float* sgcn2_b  = (const float*)d_in[16];
    const float* cgcn_w   = (const float*)d_in[17];
    const float* cgcn_b   = (const float*)d_in[18];
    const float* att1_w   = (const float*)d_in[19];
    const float* att1_b   = (const float*)d_in[20];
    const float* att2_w   = (const float*)d_in[21];
    const float* mlp_w    = (const float*)d_in[22];
    const float* mlp_b    = (const float*)d_in[23];

    char* base = (char*)d_ws;
    size_t off = 0;
    auto grab = [&](size_t bytes) { char* p = base + off; off = (off + bytes + 255) & ~(size_t)255; return p; };
    _Float16* xs16    = (_Float16*)grab((size_t)NROW * 64 * 2);
    _Float16* h16     = (_Float16*)grab((size_t)NNODE * 128 * 2);
    _Float16* fcnnW16 = (_Float16*)grab(33280 * 2);
    _Float16* lin1W16 = (_Float16*)grab(4608 * 2);
    _Float16* wih16   = (_Float16*)grab(12288 * 2);
    _Float16* whh16   = (_Float16*)grab(12288 * 2);
    _Float16* gcnW16  = (_Float16*)grab(18432 * 2);
    float*    deg     = (float*)grab((size_t)2 * NNODE * 4);
    float*    dinv    = (float*)grab((size_t)2 * NNODE * 4);
    float*    xw      = (float*)grab((size_t)3 * NNODE * 40 * 4);
    float*    acc     = (float*)grab((size_t)4 * NNODE * 40 * 4);

    float* out        = (float*)d_out;
    float* out_output = out;                         // (N,1)
    float* out_beta   = out + NNODE;                 // (N,3)
    float* out_emb1   = out + NNODE + 3 * NNODE;     // 4 x (N,40) contiguous
    float* out_emb    = out_emb1 + (size_t)4 * NNODE * 40;

    prep_kernel<<<128, 256, 0, stream>>>(fcnn_w, lin1_w, gru_wih, gru_whh,
                                         sgcn1_w, cgcn_w, sgcn2_w,
                                         fcnnW16, lin1W16, wih16, whh16, gcnW16);
    fc_kernel<<<NROW / 16, 32, 0, stream>>>(x, conv_w, conv_b, fcnnW16, fcnn_b,
                                            lin1W16, lin1_b, xs16);
    y_kernel<<<(NROW + 255) / 256, 256, 0, stream>>>(x, xs16);
    gru_kernel<<<NNODE / 16, 32, 0, stream>>>(xs16, wih16, whh16, gru_bih, gru_bhh, h16);
    h0_kernel<<<(NNODE * 64 + 255) / 256, 256, 0, stream>>>(xs16, h16);

    fill_kernel<<<(2 * NNODE + 255) / 256, 256, 0, stream>>>(deg, 1.0f, 2 * NNODE);
    deg_edge_kernel<<<(NEDGE + 255) / 256, 256, 0, stream>>>(ei + NEDGE, deg);
    deg_edge_kernel<<<(NEDGE + 255) / 256, 256, 0, stream>>>(fei + NEDGE, deg + NNODE);
    dinv_kernel<<<(2 * NNODE + 255) / 256, 256, 0, stream>>>(deg, dinv, 2 * NNODE);

    dim3 g(NNODE / 16, 3);
    xw_kernel<<<g, 32, 0, stream>>>(h16, gcnW16, xw);

    acc_init_kernel<<<(4 * NNODE * 40 + 255) / 256, 256, 0, stream>>>(xw, dinv, acc);
    // edge set 0 -> emb1 (xw set0), com1 (xw set1)
    edge_agg_kernel<<<(NEDGE + 255) / 256, 256, 0, stream>>>(
        ei, ei + NEDGE, dinv,
        xw, xw + (size_t)NNODE * 40,
        acc, acc + (size_t)NNODE * 40);
    // edge set 1 -> com2 (xw set1), emb2 (xw set2)
    edge_agg_kernel<<<(NEDGE + 255) / 256, 256, 0, stream>>>(
        fei, fei + NEDGE, dinv + NNODE,
        xw + (size_t)NNODE * 40, xw + (size_t)2 * NNODE * 40,
        acc + (size_t)2 * NNODE * 40, acc + (size_t)3 * NNODE * 40);

    bias_relu_kernel<<<(4 * NNODE * 40 + 255) / 256, 256, 0, stream>>>(
        acc, sgcn1_b, cgcn_b, sgcn2_b, out_emb1);

    att_kernel<<<(NNODE + 255) / 256, 256, 0, stream>>>(
        out_emb1,
        out_emb1 + (size_t)NNODE * 40,      // com1
        out_emb1 + (size_t)2 * NNODE * 40,  // com2
        out_emb1 + (size_t)3 * NNODE * 40,  // emb2
        att1_w, att1_b, att2_w, mlp_w, mlp_b,
        out_output, out_beta, out_emb);
}
